// NeuralFactorizationMachine_35055523070100
// MI455X (gfx1250) — compile-verified, matrix operand order checked by reading, not run on hardware
//
#include <hip/hip_runtime.h>
#include <hip/hip_bf16.h>
#include <stdint.h>

// ---------------------------------------------------------------------------
// NFM forward on MI455X (gfx1250, wave32).
//   sum_emb = X@emb ; sum_sq = X.^2 @ emb.^2 ; iv = .5*(sum_emb^2 - sum_sq)
//   h = relu(iv@w1^T + b1) ; inter = h@w2^T + b2 ; lin = X@lin_w^T + lin_b
//   out tuple = (lin+inter, lin, inter)
// All GEMMs run on V_WMMA_F32_16X16X32_F16 with f32 accumulation; operands are
// packed once to f16 in d_ws (X+lin_w = 160MB -> L2-resident on 192MB L2).
// LDS K-tiles are double-buffered via global_load_async_to_lds_b128/ASYNCcnt.
// ---------------------------------------------------------------------------

typedef __attribute__((ext_vector_type(16))) _Float16 v16h;
typedef __attribute__((ext_vector_type(8)))  _Float16 v8h;
typedef __attribute__((ext_vector_type(8)))  float    v8f;
typedef __attribute__((ext_vector_type(4)))  float    v4f;

#define DEV __device__ __forceinline__

constexpr int B_ = 1024, F_ = 16384, D_ = 256, O_ = 4096;
constexpr int KT = 32;                 // K elements per LDS stage (f16)

// ---- CDNA5 async global->LDS copy (16B per lane) --------------------------
DEV void async_cp16(const _Float16* g, _Float16* l) {
  uint32_t loff = (uint32_t)(uintptr_t)l;   // low 32 bits of generic = LDS addr
  asm volatile("global_load_async_to_lds_b128 %0, %1, off"
               :: "v"(loff), "v"(g) : "memory");
}
DEV void wait_async0() { asm volatile("s_wait_asynccnt 0" ::: "memory"); }

// ---- WMMA helpers ---------------------------------------------------------
DEV v8f wmma(v16h a, v16h b, v8f c) {
  return __builtin_amdgcn_wmma_f32_16x16x32_f16(false, a, false, b,
                                                (short)0, c, false, false);
}

// Load a 16x32 f16 fragment from an LDS subtile stored row-major [16][KT].
// Lane layout (ISA 7.12.2): lanes 0-15 row M=lane, K 0-7 & 16-23;
// lanes 16-31 row M=lane-16, K 8-15 & 24-31.  -> two ds_load_b128 per lane.
DEV v16h frag_ld(const _Float16* tile, int lane) {
  int r  = lane & 15;
  int kh = (lane >> 4) << 3;
  const _Float16* p = tile + r * KT + kh;
  v8h lo = *(const v8h*)p;
  v8h hi = *(const v8h*)(p + 16);
  v16h a;
#pragma unroll
  for (int i = 0; i < 8; ++i) { a[i] = lo[i]; a[i + 8] = hi[i]; }
  return a;
}
DEV v16h frag_sq(v16h x) { return x * x; }   // elementwise f16 squares

// ---- tile staging: A[TM][KT] + B[TN][KT] f16, 16B chunks, all 256 threads -
template<int TM, int TN>
DEV void stage(const _Float16* __restrict__ A, int lda, int aRow,
               const _Float16* __restrict__ Bm, int ldb, int bRow,
               int k0, _Float16* la, _Float16* lb, int tid) {
  constexpr int CH = (TM + TN) * 4;       // 16B chunks (row = 32 f16 = 64B)
#pragma unroll
  for (int c = tid; c < CH; c += 256) {
    if (c < TM * 4) {
      int r = c >> 2, s = (c & 3) * 8;
      async_cp16(A + (size_t)(aRow + r) * lda + k0 + s, la + r * KT + s);
    } else {
      int c2 = c - TM * 4;
      int r = c2 >> 2, s = (c2 & 3) * 8;
      async_cp16(Bm + (size_t)(bRow + r) * ldb + k0 + s, lb + r * KT + s);
    }
  }
}

// ---- core: acc += A[aRow..+128][0..K) * B[bRow..+128][0..K)^T -------------
// A: [M][K] f16 row-major, B: [N][K] f16 row-major (k-contiguous both).
template<int TM, int TN, int MS, int NS>
DEV void gemm_acc(const _Float16* __restrict__ A, int lda, int aRow,
                  const _Float16* __restrict__ Bm, int ldb, int bRow,
                  int K, _Float16* lds, v8f (&acc)[MS][NS],
                  int m0, int n0, int lane, int tid) {
  constexpr int ASZ = TM * KT, BUF = (TM + TN) * KT;
  int nk = K / KT;
  stage<TM, TN>(A, lda, aRow, Bm, ldb, bRow, 0, lds, lds + ASZ, tid);
  for (int kc = 0; kc < nk; ++kc) {
    wait_async0();
    __syncthreads();
    int cur = (kc & 1) ? BUF : 0;
    if (kc + 1 < nk)
      stage<TM, TN>(A, lda, aRow, Bm, ldb, bRow, (kc + 1) * KT,
                    lds + (BUF - cur), lds + (BUF - cur) + ASZ, tid);
    const _Float16* la = lds + cur;
    const _Float16* lb = lds + cur + ASZ;
    v16h af[MS], bf[NS];
#pragma unroll
    for (int i = 0; i < MS; ++i) af[i] = frag_ld(la + (m0 + i * 16) * KT, lane);
#pragma unroll
    for (int j = 0; j < NS; ++j) bf[j] = frag_ld(lb + (n0 + j * 16) * KT, lane);
#pragma unroll
    for (int i = 0; i < MS; ++i)
#pragma unroll
      for (int j = 0; j < NS; ++j) acc[i][j] = wmma(af[i], bf[j], acc[i][j]);
    __syncthreads();
  }
}

// ---------------------------------------------------------------------------
// pack f32 -> f16 (n multiple of 8)
__global__ void __launch_bounds__(256)
k_pack(const float* __restrict__ s, _Float16* __restrict__ d, int n) {
  int i = (blockIdx.x * 256 + threadIdx.x) * 8;
  if (i >= n) return;
  v4f x0 = *(const v4f*)(s + i);
  v4f x1 = *(const v4f*)(s + i + 4);
  v8h h;
#pragma unroll
  for (int k = 0; k < 4; ++k) { h[k] = (_Float16)x0[k]; h[k + 4] = (_Float16)x1[k]; }
  *(v8h*)(d + i) = h;
}

// emb [F][D] f32 -> embT [D][F] f16 (32x32 LDS tiles)
__global__ void __launch_bounds__(256)
k_pack_t(const float* __restrict__ e, _Float16* __restrict__ et) {
  __shared__ _Float16 t[32][33];
  int f0 = blockIdx.x * 32, d0 = blockIdx.y * 32;
  int c = threadIdx.x & 31, r8 = threadIdx.x >> 5;
#pragma unroll
  for (int i = 0; i < 4; ++i) {
    int r = r8 * 4 + i;
    t[r][c] = (_Float16)e[(size_t)(f0 + r) * D_ + d0 + c];
  }
  __syncthreads();
#pragma unroll
  for (int i = 0; i < 4; ++i) {
    int r = r8 * 4 + i;
    et[(size_t)(d0 + r) * F_ + f0 + c] = t[c][r];
  }
}

// ---------------------------------------------------------------------------
// iv = .5*((X@emb)^2 - (X.^2 @ emb.^2))  -> f16 [B][D]
__global__ void __launch_bounds__(256)
k_interaction(const _Float16* __restrict__ Xh, const _Float16* __restrict__ EtH,
              _Float16* __restrict__ ivh) {
  __shared__ _Float16 lds[2 * 2 * 128 * KT];   // 32 KB, double buffered
  constexpr int ASZ = 128 * KT, BUF = 2 * ASZ;
  int tid = threadIdx.x, lane = tid & 31, wid = tid >> 5;
  int m0 = (wid & 3) * 32, n0 = (wid >> 2) * 64;
  int gRow = blockIdx.y * 128, gCol = blockIdx.x * 128;

  v8f z = {0.f, 0.f, 0.f, 0.f, 0.f, 0.f, 0.f, 0.f};
  v8f a1[2][4], a2[2][4];
#pragma unroll
  for (int i = 0; i < 2; ++i)
#pragma unroll
    for (int j = 0; j < 4; ++j) { a1[i][j] = z; a2[i][j] = z; }

  int nk = F_ / KT;
  stage<128, 128>(Xh, F_, gRow, EtH, F_, gCol, 0, lds, lds + ASZ, tid);
  for (int kc = 0; kc < nk; ++kc) {
    wait_async0();
    __syncthreads();
    int cur = (kc & 1) ? BUF : 0;
    if (kc + 1 < nk)
      stage<128, 128>(Xh, F_, gRow, EtH, F_, gCol, (kc + 1) * KT,
                      lds + (BUF - cur), lds + (BUF - cur) + ASZ, tid);
    const _Float16* la = lds + cur;
    const _Float16* lb = lds + cur + ASZ;
    v16h af[2], bf[4], af2[2], bf2[4];
#pragma unroll
    for (int i = 0; i < 2; ++i) {
      af[i]  = frag_ld(la + (m0 + i * 16) * KT, lane);
      af2[i] = frag_sq(af[i]);
    }
#pragma unroll
    for (int j = 0; j < 4; ++j) {
      bf[j]  = frag_ld(lb + (n0 + j * 16) * KT, lane);
      bf2[j] = frag_sq(bf[j]);
    }
#pragma unroll
    for (int i = 0; i < 2; ++i)
#pragma unroll
      for (int j = 0; j < 4; ++j) {
        a1[i][j] = wmma(af[i],  bf[j],  a1[i][j]);
        a2[i][j] = wmma(af2[i], bf2[j], a2[i][j]);
      }
    __syncthreads();
  }
  // epilogue: iv = .5*(s*s - q)
#pragma unroll
  for (int i = 0; i < 2; ++i)
#pragma unroll
    for (int j = 0; j < 4; ++j) {
      int c  = gCol + n0 + j * 16 + (lane & 15);
      int r0 = gRow + m0 + i * 16 + ((lane >> 4) << 3);
#pragma unroll
      for (int e = 0; e < 8; ++e) {
        float s = a1[i][j][e], q = a2[i][j][e];
        ivh[(size_t)(r0 + e) * D_ + c] = (_Float16)(0.5f * (s * s - q));
      }
    }
}

// h = relu(iv @ w1^T + b1) -> f16 [B][D]
__global__ void __launch_bounds__(256)
k_mlp1(const _Float16* __restrict__ ivh, const _Float16* __restrict__ W1h,
       const float* __restrict__ b1, _Float16* __restrict__ hh) {
  __shared__ _Float16 lds[2 * 2 * 128 * KT];
  int tid = threadIdx.x, lane = tid & 31, wid = tid >> 5;
  int m0 = (wid & 3) * 32, n0 = (wid >> 2) * 64;
  int gRow = blockIdx.y * 128, gCol = blockIdx.x * 128;

  v8f z = {0.f, 0.f, 0.f, 0.f, 0.f, 0.f, 0.f, 0.f};
  v8f acc[2][4];
#pragma unroll
  for (int i = 0; i < 2; ++i)
#pragma unroll
    for (int j = 0; j < 4; ++j) acc[i][j] = z;

  gemm_acc<128, 128, 2, 4>(ivh, D_, gRow, W1h, D_, gCol, D_, lds, acc,
                           m0, n0, lane, tid);
#pragma unroll
  for (int i = 0; i < 2; ++i)
#pragma unroll
    for (int j = 0; j < 4; ++j) {
      int c  = gCol + n0 + j * 16 + (lane & 15);
      int r0 = gRow + m0 + i * 16 + ((lane >> 4) << 3);
      float bias = b1[c];
#pragma unroll
      for (int e = 0; e < 8; ++e) {
        float v = acc[i][j][e] + bias;
        hh[(size_t)(r0 + e) * D_ + c] = (_Float16)(v > 0.f ? v : 0.f);
      }
    }
}

// lin = X@lin_w^T + lin_b ; inter = h@w2^T + b2 ; out = lin + inter
// d_out = [output | linear_out | interaction_out], each [B][O] f32
__global__ void __launch_bounds__(256)
k_output(const _Float16* __restrict__ Xh, const _Float16* __restrict__ Wh,
         const float* __restrict__ lin_b,
         const _Float16* __restrict__ hh, const _Float16* __restrict__ W2h,
         const float* __restrict__ b2, float* __restrict__ out) {
  __shared__ _Float16 lds[2 * 2 * 128 * KT];
  int tid = threadIdx.x, lane = tid & 31, wid = tid >> 5;
  int m0 = (wid & 3) * 32, n0 = (wid >> 2) * 64;
  int gRow = blockIdx.y * 128, gCol = blockIdx.x * 128;
  const size_t BO = (size_t)B_ * O_;
  float* lin   = out + BO;
  float* inter = out + 2 * BO;

  v8f z = {0.f, 0.f, 0.f, 0.f, 0.f, 0.f, 0.f, 0.f};
  v8f acc[2][4];
#pragma unroll
  for (int i = 0; i < 2; ++i)
#pragma unroll
    for (int j = 0; j < 4; ++j) acc[i][j] = z;

  // phase 1: the 137-GFLOP GEMM over F
  gemm_acc<128, 128, 2, 4>(Xh, F_, gRow, Wh, F_, gCol, F_, lds, acc,
                           m0, n0, lane, tid);
#pragma unroll
  for (int i = 0; i < 2; ++i)
#pragma unroll
    for (int j = 0; j < 4; ++j) {
      int c  = gCol + n0 + j * 16 + (lane & 15);
      int r0 = gRow + m0 + i * 16 + ((lane >> 4) << 3);
      float bias = lin_b[c];
#pragma unroll
      for (int e = 0; e < 8; ++e)
        lin[(size_t)(r0 + e) * O_ + c] = acc[i][j][e] + bias;
    }
  __syncthreads();

  // phase 2: interaction head over D (reuse LDS + accumulators)
#pragma unroll
  for (int i = 0; i < 2; ++i)
#pragma unroll
    for (int j = 0; j < 4; ++j) acc[i][j] = z;
  gemm_acc<128, 128, 2, 4>(hh, D_, gRow, W2h, D_, gCol, D_, lds, acc,
                           m0, n0, lane, tid);
#pragma unroll
  for (int i = 0; i < 2; ++i)
#pragma unroll
    for (int j = 0; j < 4; ++j) {
      int c  = gCol + n0 + j * 16 + (lane & 15);
      int r0 = gRow + m0 + i * 16 + ((lane >> 4) << 3);
      float bias = b2[c];
#pragma unroll
      for (int e = 0; e < 8; ++e) {
        size_t idx = (size_t)(r0 + e) * O_ + c;
        float it = acc[i][j][e] + bias;
        inter[idx] = it;
        out[idx]   = lin[idx] + it;
      }
    }
}

// ---------------------------------------------------------------------------
extern "C" void kernel_launch(void* const* d_in, const int* in_sizes, int n_in,
                              void* d_out, int out_size, void* d_ws, size_t ws_size,
                              hipStream_t stream) {
  (void)in_sizes; (void)n_in; (void)out_size; (void)ws_size;
  const float* sae   = (const float*)d_in[0];
  const float* emb   = (const float*)d_in[1];
  const float* lin_w = (const float*)d_in[2];
  const float* lin_b = (const float*)d_in[3];
  const float* w1    = (const float*)d_in[4];
  const float* b1    = (const float*)d_in[5];
  const float* w2    = (const float*)d_in[6];
  const float* b2    = (const float*)d_in[7];
  float* out = (float*)d_out;

  _Float16* ws = (_Float16*)d_ws;           // f16 element offsets
  _Float16* Xh  = ws;                                       // B*F
  _Float16* Wh  = Xh  + (size_t)B_ * F_;                    // O*F
  _Float16* EtH = Wh  + (size_t)O_ * F_;                    // D*F (emb^T)
  _Float16* W1h = EtH + (size_t)D_ * F_;                    // D*D
  _Float16* W2h = W1h + (size_t)D_ * D_;                    // O*D
  _Float16* IVh = W2h + (size_t)O_ * D_;                    // B*D
  _Float16* Hh  = IVh + (size_t)B_ * D_;                    // B*D

  k_pack<<<(B_ * F_) / 2048, 256, 0, stream>>>(sae,   Xh,  B_ * F_);
  k_pack<<<(O_ * F_) / 2048, 256, 0, stream>>>(lin_w, Wh,  O_ * F_);
  k_pack<<<(D_ * D_) / 2048, 256, 0, stream>>>(w1,    W1h, D_ * D_);
  k_pack<<<(O_ * D_) / 2048, 256, 0, stream>>>(w2,    W2h, O_ * D_);
  k_pack_t<<<dim3(F_ / 32, D_ / 32), 256, 0, stream>>>(emb, EtH);

  k_interaction<<<dim3(D_ / 128, B_ / 128), 256, 0, stream>>>(Xh, EtH, IVh);
  k_mlp1<<<dim3(D_ / 128, B_ / 128), 256, 0, stream>>>(IVh, W1h, b1, Hh);
  k_output<<<dim3(O_ / 128, B_ / 128), 256, 0, stream>>>(Xh, Wh, lin_b,
                                                         Hh, W2h, b2, out);
}
